// RBF_62002147885585
// MI455X (gfx1250) — matrix operand order repaired and checked by reading, not compiled, not run on hardware
//
#include <hip/hip_runtime.h>
#include <hip/hip_fp16.h>

typedef __attribute__((ext_vector_type(16))) _Float16 v16h;
typedef __attribute__((ext_vector_type(8)))  _Float16 v8h;
typedef __attribute__((ext_vector_type(8)))  float    v8f;

#define M_TOTAL 32768   // B * L = 8 * 4096
#define N_TOTAL 1024    // D (centers)
#define K_TOTAL 1024    // D (feature dim)
#define BM 128
#define BN 128
#define BK 32
#define NK (K_TOTAL / BK)   // 32 K-tiles
#define LDT 40              // padded LDS row stride in halves (80B: keeps b128 aligned, spreads banks)

__global__ __launch_bounds__(256)
void rbf_wmma_kernel(const float* __restrict__ X,      // (M, K) row-major
                     const float* __restrict__ C,      // (K, N) row-major
                     const float* __restrict__ W,      // (N,)
                     float* __restrict__ Out) {        // (M, N) row-major
  __shared__ _Float16 sA[2][BM * LDT];
  __shared__ _Float16 sB[2][BN * LDT];
  __shared__ float sRow[BM];   // sum_k x[m,k]^2 for rows of this block
  __shared__ float sCol[BN];   // sum_k c[k,n]^2 for cols of this block

  const int t    = threadIdx.x;
  const int lane = t & 31;
  const int wave = t >> 5;          // 0..7
  const int wm   = wave >> 2;       // 0..1 : 64-row slab
  const int wn   = wave & 3;        // 0..3 : 32-col slab

  const int bm = blockIdx.y;        // 0..255 (M blocks)
  const int bn = blockIdx.x;        // 0..7   (N blocks, fastest -> x panel L2 reuse)

  if (t < BM) sRow[t] = 0.f;
  if (t < BN) sCol[t] = 0.f;

  // Staging assignment: each thread owns one A row-halfchunk and one B column-halfchunk.
  const int ar  = t >> 1;           // A local row 0..127
  const int ac  = (t & 1) * 16;     // A k-chunk base {0,16}
  const int bnl = t >> 1;           // B local col (n) 0..127
  const int bk0 = (t & 1) * 16;     // B k-chunk base {0,16}

  const float* xbase = X + (size_t)(bm * BM + ar) * K_TOTAL + ac;
  const float* cbase = C + (size_t)bk0 * N_TOTAL + (size_t)bn * BN + bnl;

  float rowAcc = 0.f, colAcc = 0.f;
  float ra[16], rb[16];

  // Global -> registers for K-tile kt (also accumulates squared norms for free).
  auto gload = [&](int kt) {
    const float4* xp = (const float4*)(xbase + kt * BK);
#pragma unroll
    for (int j = 0; j < 4; ++j) {
      float4 v = xp[j];
      ra[4*j+0] = v.x; ra[4*j+1] = v.y; ra[4*j+2] = v.z; ra[4*j+3] = v.w;
    }
#pragma unroll
    for (int i = 0; i < 16; ++i) rowAcc += ra[i] * ra[i];

    const float* cp = cbase + (size_t)kt * BK * N_TOTAL;
#pragma unroll
    for (int i = 0; i < 16; ++i) {
      float v = cp[(size_t)i * N_TOTAL];
      rb[i] = v;
      colAcc += v * v;
    }
    if (kt + 4 < NK) {  // global_prefetch_b8 four tiles ahead
      __builtin_prefetch(xbase + (size_t)(kt + 4) * BK, 0, 0);
      __builtin_prefetch(cbase + (size_t)(kt + 4) * BK * N_TOTAL, 0, 0);
    }
  };

  // Registers -> LDS (f32 -> f16 conversion here, done once per element).
  // A is stored row-major [m][k]; B is stored TRANSPOSED [n][k] so B-fragment
  // reads (16 contiguous K per lane) are two aligned ds_read_b128.
  auto stage = [&](int buf) {
    v8h lo, hi;
#pragma unroll
    for (int i = 0; i < 8; ++i) { lo[i] = (_Float16)ra[i]; hi[i] = (_Float16)ra[i+8]; }
    *(v8h*)&sA[buf][ar * LDT + ac]      = lo;
    *(v8h*)&sA[buf][ar * LDT + ac + 8]  = hi;
#pragma unroll
    for (int i = 0; i < 8; ++i) { lo[i] = (_Float16)rb[i]; hi[i] = (_Float16)rb[i+8]; }
    *(v8h*)&sB[buf][bnl * LDT + bk0]     = lo;
    *(v8h*)&sB[buf][bnl * LDT + bk0 + 8] = hi;
  };

  // A 16x32 layout (matches B): lane<16 -> row=lane, K 0..15;
  //                            lane>=16 -> row=lane-16, K 16..31.
  auto loadA = [&](int buf, int mi) -> v16h {
    int row = wm * 64 + mi * 16 + (lane & 15);
    int k0  = (lane < 16) ? 0 : 16;
    const _Float16* p = &sA[buf][row * LDT + k0];
    v8h lo = *(const v8h*)p;
    v8h hi = *(const v8h*)(p + 8);
    return __builtin_shufflevector(lo, hi, 0,1,2,3,4,5,6,7,8,9,10,11,12,13,14,15);
  };
  // B 32x16: lane<16 -> N=lane, K 0..15; lane>=16 -> N=lane-16, K 16..31.
  auto loadB = [&](int buf, int ni) -> v16h {
    int col = wn * 32 + ni * 16 + (lane & 15);
    int k0  = (lane < 16) ? 0 : 16;
    const _Float16* p = &sB[buf][col * LDT + k0];
    v8h lo = *(const v8h*)p;
    v8h hi = *(const v8h*)(p + 8);
    return __builtin_shufflevector(lo, hi, 0,1,2,3,4,5,6,7,8,9,10,11,12,13,14,15);
  };

  v8f acc[4][2];
  v8f zero = {};
#pragma unroll
  for (int mi = 0; mi < 4; ++mi)
#pragma unroll
    for (int ni = 0; ni < 2; ++ni) acc[mi][ni] = zero;

  // Software pipeline: regs hold tile kt+1 at top of iteration kt.
  gload(0);
  stage(0);
  gload(1);
  __syncthreads();

  for (int kt = 0; kt < NK; ++kt) {
    const int cur = kt & 1;
    if (kt + 1 < NK) stage(cur ^ 1);     // write tile kt+1 into the other buffer
    if (kt + 2 < NK) gload(kt + 2);      // issue global loads for tile kt+2

    v16h bfrag0 = loadB(cur, 0);
    v16h bfrag1 = loadB(cur, 1);
#pragma unroll
    for (int mi = 0; mi < 4; ++mi) {
      v16h a = loadA(cur, mi);
      acc[mi][0] = __builtin_amdgcn_wmma_f32_16x16x32_f16(
          false, a, false, bfrag0, (short)0, acc[mi][0], false, false);
      acc[mi][1] = __builtin_amdgcn_wmma_f32_16x16x32_f16(
          false, a, false, bfrag1, (short)0, acc[mi][1], false, false);
    }
    __syncthreads();
  }

  // Merge per-thread squared-norm partials (2 threads per row / per col).
  atomicAdd(&sRow[ar],  rowAcc);
  atomicAdd(&sCol[bnl], colAcc);
  __syncthreads();

  // Epilogue. C/D fragment layout: VGPR r -> M = r + 8*(lane>=16), N = lane&15.
  const int nl15 = lane & 15;
  const int rofs = (lane >> 4) * 8;
#pragma unroll
  for (int ni = 0; ni < 2; ++ni) {
    const int n_l = wn * 32 + ni * 16 + nl15;
    const int n_g = bn * BN + n_l;
    const float c2 = sCol[n_l];
    const float w  = W[n_g];
    const float negInv = -1.0f / (2.0f * w * w);
#pragma unroll
    for (int mi = 0; mi < 4; ++mi) {
#pragma unroll
      for (int r = 0; r < 8; ++r) {
        const int m_l = wm * 64 + mi * 16 + rofs + r;
        const float v  = acc[mi][ni][r];
        const float sq = sRow[m_l] - 2.0f * v + c2;
        Out[(size_t)(bm * BM + m_l) * N_TOTAL + n_g] = __expf(sq * negInv);
      }
    }
  }
}

extern "C" void kernel_launch(void* const* d_in, const int* in_sizes, int n_in,
                              void* d_out, int out_size, void* d_ws, size_t ws_size,
                              hipStream_t stream) {
  const float* x = (const float*)d_in[0];   // (8, 4096, 1024) f32
  const float* c = (const float*)d_in[1];   // (1024, 1024) f32
  const float* w = (const float*)d_in[2];   // (1024,) f32
  float* out = (float*)d_out;               // (8, 4096, 1024) f32

  dim3 grid(N_TOTAL / BN, M_TOTAL / BM);    // (8, 256): n fastest for x L2 reuse
  dim3 block(256);
  rbf_wmma_kernel<<<grid, block, 0, stream>>>(x, c, w, out);
}
